// CausalSelfAttention_59098749993568
// MI455X (gfx1250) — compile-verified, matrix operand order checked by reading, not run on hardware
//
#include <hip/hip_runtime.h>
#include <hip/hip_bf16.h>

#define B_    2
#define S_    2048
#define HDIM  1024
#define NH    16
#define HD    64
#define BS    (B_ * S_)
#define CHUNK 64
#define EPSL  1e-6f

typedef _Float16 half_t;
typedef __attribute__((ext_vector_type(16))) _Float16 v16h;
typedef __attribute__((ext_vector_type(8)))  float    v8f;
typedef int v4i_t __attribute__((vector_size(16)));   // matches async-LDS builtin params

#if defined(__gfx1250__) && \
    __has_builtin(__builtin_amdgcn_global_load_async_to_lds_b128) && \
    __has_builtin(__builtin_amdgcn_s_wait_asynccnt)
#define USE_ASYNC_LDS 1
#endif

__device__ inline v8f wmma16(v16h a, v16h b, v8f c) {
  // D = A(16x32 f16) * B(32x16 f16) + C(16x16 f32)
  return __builtin_amdgcn_wmma_f32_16x16x32_f16(
      /*neg_a=*/false, a, /*neg_b=*/false, b,
      /*c_mod=*/(short)0, c, /*reuse_a=*/false, /*reuse_b=*/false);
}

// A operand: 16x32 tile, element [m][k] = src[(row0+m)*ld + k0+k]
// Lane layout (ISA 7.12.2): lanes 0-15 rows, halves 0..7 -> K 0..7, 8..15 -> K 16..23;
// lanes 16-31 same rows with K base +8.
template <typename T>
__device__ inline v16h ld_a(const T* src, int row0, int ld, int k0) {
  const int lane = threadIdx.x & 31;
  const int r  = row0 + (lane & 15);
  const int kb = k0 + ((lane & 16) >> 1);     // +8 for upper half-wave
  const T* p = src + (size_t)r * ld + kb;
  v16h out;
#pragma unroll
  for (int j = 0; j < 8; ++j) {
    out[j]     = (half_t)p[j];
    out[j + 8] = (half_t)p[16 + j];
  }
  return out;
}

// B operand where B[k][n] = srcT[n][k] (k contiguous): same access pattern as A.
template <typename T>
__device__ inline v16h ld_bt(const T* src, int col0, int ld, int k0) {
  return ld_a(src, col0, ld, k0);
}

// B operand where B[k][n] = src[k*ld + n] (k strided).
template <typename T>
__device__ inline v16h ld_b(const T* src, int col0, int ld, int k0) {
  const int lane = threadIdx.x & 31;
  const int n  = col0 + (lane & 15);
  const int kb = k0 + ((lane & 16) >> 1);
  const T* p = src + (size_t)kb * ld + n;
  v16h out;
#pragma unroll
  for (int j = 0; j < 8; ++j) {
    out[j]     = (half_t)p[(size_t)j * ld];
    out[j + 8] = (half_t)p[(size_t)(16 + j) * ld];
  }
  return out;
}

// ---------------- QKV projection: Y = X @ W^T + b, optional phi ----------------
// output written f16 in (b, h, s, d) head-major layout for the attention kernel.
__global__ __launch_bounds__(32) void k_qkv_proj(
    const float* __restrict__ X, const float* __restrict__ W,
    const float* __restrict__ bias, half_t* __restrict__ out, int apply_phi)
{
  const int m0 = blockIdx.y * 16;   // row strip over BS=4096
  const int n0 = blockIdx.x * 64;   // 4 N-tiles over 1024
  v8f cc[4] = {};
  for (int k0 = 0; k0 < HDIM; k0 += 32) {
    v16h a = ld_a(X, m0, HDIM, k0);             // fp32 -> f16 convert on load
    v16h bt[4];
#pragma unroll
    for (int t = 0; t < 4; ++t)
      bt[t] = ld_bt(W, n0 + 16 * t, HDIM, k0);  // B[k][n] = W[n][k], contiguous
#pragma unroll
    for (int t = 0; t < 4; ++t)
      cc[t] = wmma16(a, bt[t], cc[t]);
  }
  const int lane  = threadIdx.x & 31;
  const int rbase = m0 + ((lane & 16) >> 1);    // C layout: +8 rows upper half-wave
#pragma unroll
  for (int t = 0; t < 4; ++t) {
    const int col = n0 + 16 * t + (lane & 15);
    const float bcol = bias[col];
    const int h = col >> 6, d = col & 63;
#pragma unroll
    for (int r = 0; r < 8; ++r) {
      const int row = rbase + r;
      float y = cc[t][r] + bcol;
      if (apply_phi) y = (y > 0.f) ? (y + 1.f) : __expf(y);   // elu(y)+1
      const int bb = row >> 11;           // row / S_
      const int s  = row & (S_ - 1);
      out[(((size_t)(bb * NH + h)) * S_ + s) * HD + d] = (half_t)y;
    }
  }
}

// ---------------- Chunked causal linear attention, one block per (b,h) ----------------
__global__ __launch_bounds__(256) void k_linear_attn(
    const half_t* __restrict__ qphi, const half_t* __restrict__ kphi,
    const half_t* __restrict__ vv, half_t* __restrict__ ao)
{
  __shared__ float  Sst[HD * HD];        // running k^T v state (f32)
  __shared__ float  zst[HD];             // running sum of phi(k)
  __shared__ float  den[CHUNK];          // per-row denominator
  __shared__ __align__(16) half_t qc[CHUNK * HD];
  __shared__ __align__(16) half_t kc[CHUNK * HD];
  __shared__ __align__(16) half_t vc[CHUNK * HD];
  __shared__ half_t Am[CHUNK * CHUNK];   // masked scores (f16)

  const int tid  = threadIdx.x;
  const int lane = tid & 31;
  const int w    = tid >> 5;             // 8 waves
  const int bh   = blockIdx.x;
  const int b    = bh >> 4;
  const int h    = bh & 15;
  const size_t hbase = (size_t)bh * S_ * HD;

  for (int i = tid; i < HD * HD; i += 256) Sst[i] = 0.f;
  if (tid < HD) zst[tid] = 0.f;
  __syncthreads();

  for (int c = 0; c < S_ / CHUNK; ++c) {
    const size_t cbase = hbase + (size_t)c * CHUNK * HD;

#ifdef USE_ASYNC_LDS
    // memory -> LDS without the VGPR round-trip; tracked by ASYNCcnt
    for (int i = tid * 8; i < CHUNK * HD; i += 256 * 8) {   // 16B granules
      __builtin_amdgcn_global_load_async_to_lds_b128(
          (v4i_t*)(qphi + cbase + i), (v4i_t*)(qc + i), 0, 0);
      __builtin_amdgcn_global_load_async_to_lds_b128(
          (v4i_t*)(kphi + cbase + i), (v4i_t*)(kc + i), 0, 0);
      __builtin_amdgcn_global_load_async_to_lds_b128(
          (v4i_t*)(vv + cbase + i), (v4i_t*)(vc + i), 0, 0);
    }
#else
    for (int i = tid; i < CHUNK * HD; i += 256) {
      qc[i] = qphi[cbase + i];
      kc[i] = kphi[cbase + i];
      vc[i] = vv[cbase + i];
    }
#endif
    // warm next chunk (128B per lane) while this chunk is consumed
    if (c + 1 < S_ / CHUNK) {
      const size_t nb = cbase + (size_t)CHUNK * HD;
      if (tid < 64)       __builtin_prefetch(qphi + nb + (size_t)tid * 64, 0, 0);
      else if (tid < 128) __builtin_prefetch(kphi + nb + (size_t)(tid - 64) * 64, 0, 0);
      else if (tid < 192) __builtin_prefetch(vv  + nb + (size_t)(tid - 128) * 64, 0, 0);
    }
    if (tid < CHUNK) den[tid] = 0.f;
#ifdef USE_ASYNC_LDS
    __builtin_amdgcn_s_wait_asynccnt(0);
#endif
    __syncthreads();

    // inter-chunk denominator: phi(q)_i . z_state
    if (tid < CHUNK) {
      float acc = 0.f;
      for (int d = 0; d < HD; ++d) acc += (float)qc[tid * HD + d] * zst[d];
      atomicAdd(&den[tid], acc);
    }

    v8f onum[2];
#pragma unroll
    for (int tt = 0; tt < 2; ++tt) {            // each wave owns 2 of 16 tiles
      const int t  = w + tt * 8;
      const int tm = (t >> 2) * 16, tn = (t & 3) * 16;
      v8f sc = {};   // scores  qc @ kc^T
      v8f on = {};   // inter   qc @ Sst
#pragma unroll
      for (int k0 = 0; k0 < HD; k0 += 32) {
        v16h a = ld_a(qc, tm, HD, k0);
        sc = wmma16(a, ld_bt(kc, tn, HD, k0), sc);
        on = wmma16(a, ld_b(Sst, tn, HD, k0), on);   // f32 LDS -> f16 operand
      }
      const int coln = tn + (lane & 15);
      const int rb   = tm + ((lane & 16) >> 1);
#pragma unroll
      for (int r = 0; r < 8; ++r) {
        const int row = rb + r;
        const float mv = (row >= coln) ? sc[r] : 0.f;   // causal mask (inclusive)
        Am[row * CHUNK + coln] = (half_t)mv;
        atomicAdd(&den[row], mv);                       // intra denominator row-sum
      }
      onum[tt] = on;
    }
    __syncthreads();   // Am/den ready; all Sst reads done

#pragma unroll
    for (int tt = 0; tt < 2; ++tt) {
      const int t  = w + tt * 8;
      const int tm = (t >> 2) * 16, tn = (t & 3) * 16;
      v8f on = onum[tt];
#pragma unroll
      for (int k0 = 0; k0 < CHUNK; k0 += 32)   // intra: += A_masked @ v
        on = wmma16(ld_a(Am, tm, CHUNK, k0), ld_b(vc, tn, HD, k0), on);

      const int coln = tn + (lane & 15);
      const int rb   = tm + ((lane & 16) >> 1);
#pragma unroll
      for (int r = 0; r < 8; ++r) {
        const int row = rb + r;
        const float o = on[r] / (den[row] + EPSL);
        const int sg  = c * CHUNK + row;
        ao[((size_t)b * S_ + sg) * HDIM + h * HD + coln] = (half_t)o;
      }

      // state update: Sst += kc^T @ vc   (A[i][k] = kc[k][i] -> strided load)
      v8f su = {};
#pragma unroll
      for (int k0 = 0; k0 < CHUNK; k0 += 32)
        su = wmma16(ld_b(kc, tm, HD, k0), ld_b(vc, tn, HD, k0), su);
#pragma unroll
      for (int r = 0; r < 8; ++r)
        Sst[(rb + r) * HD + coln] += su[r];
    }

    // z update: z += column sums of phi(k) chunk
    if (tid < HD) {
      float za = 0.f;
      for (int s = 0; s < CHUNK; ++s) za += (float)kc[s * HD + tid];
      zst[tid] += za;
    }
    __syncthreads();
  }
}

// ---------------- Output projection: out = AO @ Wo^T + bo (f32 result) ----------------
__global__ __launch_bounds__(32) void k_out_proj(
    const half_t* __restrict__ AO, const float* __restrict__ Wo,
    const float* __restrict__ bo, float* __restrict__ Y)
{
  const int m0 = blockIdx.y * 16;
  const int n0 = blockIdx.x * 64;
  v8f cc[4] = {};
  for (int k0 = 0; k0 < HDIM; k0 += 32) {
    v16h a = ld_a(AO, m0, HDIM, k0);            // already f16 in workspace
    v16h bt[4];
#pragma unroll
    for (int t = 0; t < 4; ++t)
      bt[t] = ld_bt(Wo, n0 + 16 * t, HDIM, k0);
#pragma unroll
    for (int t = 0; t < 4; ++t)
      cc[t] = wmma16(a, bt[t], cc[t]);
  }
  const int lane  = threadIdx.x & 31;
  const int rbase = m0 + ((lane & 16) >> 1);
#pragma unroll
  for (int t = 0; t < 4; ++t) {
    const int col = n0 + 16 * t + (lane & 15);
    const float bv = bo[col];
#pragma unroll
    for (int r = 0; r < 8; ++r)
      Y[(size_t)(rbase + r) * HDIM + col] = cc[t][r] + bv;
  }
}

extern "C" void kernel_launch(void* const* d_in, const int* in_sizes, int n_in,
                              void* d_out, int out_size, void* d_ws, size_t ws_size,
                              hipStream_t stream) {
  const float* query = (const float*)d_in[0];
  const float* key   = (const float*)d_in[1];
  const float* value = (const float*)d_in[2];
  const float* Wq    = (const float*)d_in[3];
  const float* bq    = (const float*)d_in[4];
  const float* Wk    = (const float*)d_in[5];
  const float* bk    = (const float*)d_in[6];
  const float* Wv    = (const float*)d_in[7];
  const float* bv    = (const float*)d_in[8];
  const float* Wo    = (const float*)d_in[9];
  const float* bo    = (const float*)d_in[10];
  float* out = (float*)d_out;

  // workspace: 4 x 8MB f16 tensors
  half_t* qphi = (half_t*)d_ws;
  half_t* kphi = qphi + (size_t)BS * HDIM;
  half_t* vh   = kphi + (size_t)BS * HDIM;
  half_t* ao   = vh   + (size_t)BS * HDIM;

  dim3 gp(HDIM / 64, BS / 16);
  k_qkv_proj<<<gp, dim3(32), 0, stream>>>(query, Wq, bq, qphi, 1);
  k_qkv_proj<<<gp, dim3(32), 0, stream>>>(key,   Wk, bk, kphi, 1);
  k_qkv_proj<<<gp, dim3(32), 0, stream>>>(value, Wv, bv, vh,   0);
  k_linear_attn<<<dim3(B_ * NH), dim3(256), 0, stream>>>(qphi, kphi, vh, ao);
  k_out_proj<<<gp, dim3(32), 0, stream>>>(ao, Wo, bo, out);
}
